// Transformer_16320875725342
// MI455X (gfx1250) — compile-verified, matrix operand order checked by reading, not run on hardware
//
#include <hip/hip_runtime.h>
#include <hip/hip_bf16.h>

#define BB   4
#define NN   1024
#define DIMD 1024
#define HH   16
#define DHH  64
#define MLPDIM 4096
#define LLAYERS 2
#define CC   256
#define MTOT (BB * NN)   // 4096

typedef __attribute__((ext_vector_type(16))) __bf16 v16bf;
typedef __attribute__((ext_vector_type(8)))  __bf16 v8bf;
typedef __attribute__((ext_vector_type(8)))  float  v8f;

// ---------------------------------------------------------------------------
// WMMA helpers (CDNA5: D = A x B + C, 16x16x32 bf16, f32 accumulate)
// ---------------------------------------------------------------------------
__device__ __forceinline__ v8f wmma_bf16(v16bf a, v16bf b, v8f c) {
  return __builtin_amdgcn_wmma_f32_16x16x32_bf16(false, a, false, b, (short)0, c,
                                                 false, false);
}

// A-fragment (and symmetric B-fragment) loader for 16-bit 16x32 tiles.
// Per ISA 7.12.2: lanes 0-15 hold K = {kb..kb+7, kb+16..kb+23} with kb=0,
// lanes 16-31 the same with kb=8.  base = row (A) / column-of-W (B=W^T row).
__device__ __forceinline__ v16bf load_frag(const __bf16* base, int kb) {
  v8bf lo = *(const v8bf*)(base + kb);
  v8bf hi = *(const v8bf*)(base + kb + 16);
  return __builtin_shufflevector(lo, hi, 0, 1, 2, 3, 4, 5, 6, 7,
                                 8, 9, 10, 11, 12, 13, 14, 15);
}

// Pure-VALU 16-lane max reduction via DPP (no LDS round-trips).
// xor1 = quad_perm(1,0,3,2); xor2 = quad_perm(2,3,0,1); then half/row mirror
// act as xor4/xor8 once quads are uniform.
template <int CTRL>
__device__ __forceinline__ float dpp_maxstep(float x) {
  int xi = __builtin_bit_cast(int, x);
  int yi = __builtin_amdgcn_update_dpp(xi, xi, CTRL, 0xF, 0xF, false);
  return fmaxf(x, __builtin_bit_cast(float, yi));
}
__device__ __forceinline__ float rowmax16(float x) {
  x = dpp_maxstep<0xB1>(x);   // quad_perm [1,0,3,2]
  x = dpp_maxstep<0x4E>(x);   // quad_perm [2,3,0,1]
  x = dpp_maxstep<0x141>(x);  // row_half_mirror
  x = dpp_maxstep<0x140>(x);  // row_mirror
  return x;
}

// ---------------------------------------------------------------------------
// Elementwise / prep kernels
// ---------------------------------------------------------------------------
__global__ void k_silu_bf16(const float* __restrict__ in, __bf16* __restrict__ out,
                            int n) {
  int i = blockIdx.x * 256 + threadIdx.x;
  if (i < n) {
    float v = in[i];
    out[i] = (__bf16)(v / (1.f + __expf(-v)));
  }
}

// fp32 [K][Nc] row-major  ->  bf16 [Nc][K] row-major (i.e. W^T)
__global__ void k_transpose_cast(const float* __restrict__ in,
                                 __bf16* __restrict__ out, int K, int Nc) {
  __shared__ float tile[32][33];
  int k0 = blockIdx.y * 32, n0 = blockIdx.x * 32;
  int tx = threadIdx.x & 31, ty = threadIdx.x >> 5;  // 256 thr: ty 0..7
#pragma unroll
  for (int i = 0; i < 32; i += 8)
    tile[ty + i][tx] = in[(size_t)(k0 + ty + i) * Nc + n0 + tx];
  __syncthreads();
#pragma unroll
  for (int i = 0; i < 32; i += 8)
    out[(size_t)(n0 + ty + i) * K + k0 + tx] = (__bf16)tile[tx][ty + i];
}

// h = LN(x; g, be) * (1 + sc) + sh   -> bf16.  One block per token row.
__global__ void k_ln_mod(const float* __restrict__ x, const float* __restrict__ g,
                         const float* __restrict__ be, const float* __restrict__ mm,
                         const float* __restrict__ mf, int jsh, int jsc,
                         __bf16* __restrict__ out) {
  int r = blockIdx.x;
  const float* xr = x + (size_t)r * DIMD;
  float v[4], s = 0.f, ss = 0.f;
#pragma unroll
  for (int i = 0; i < 4; i++) {
    v[i] = xr[threadIdx.x + 256 * i];
    s += v[i];
    ss += v[i] * v[i];
  }
  __shared__ float red[16];
#pragma unroll
  for (int m = 16; m; m >>= 1) {
    s += __shfl_xor(s, m, 32);
    ss += __shfl_xor(ss, m, 32);
  }
  int w = threadIdx.x >> 5;
  if ((threadIdx.x & 31) == 0) { red[w] = s; red[8 + w] = ss; }
  __syncthreads();
  if (threadIdx.x == 0) {
    float S = 0.f, SS = 0.f;
#pragma unroll
    for (int i = 0; i < 8; i++) { S += red[i]; SS += red[8 + i]; }
    float mean = S * (1.f / DIMD);
    float var = SS * (1.f / DIMD) - mean * mean;
    red[0] = mean;
    red[1] = rsqrtf(var + 1e-5f);
  }
  __syncthreads();
  float mean = red[0], rstd = red[1];
  const float* mmr = mm + (size_t)r * (6 * CC);
  const float* mfr = mf + (size_t)r * (18 * CC);
#pragma unroll
  for (int i = 0; i < 4; i++) {
    int c = threadIdx.x + 256 * i;
    float sc = (c < CC) ? mmr[jsc * CC + c] : mfr[jsc * 3 * CC + (c - CC)];
    float sh = (c < CC) ? mmr[jsh * CC + c] : mfr[jsh * 3 * CC + (c - CC)];
    float hv = (v[i] - mean) * rstd * g[c] + be[c];
    out[(size_t)r * DIMD + c] = (__bf16)(hv * (1.f + sc) + sh);
  }
}

// ---------------------------------------------------------------------------
// Tiled WMMA GEMM: C[M x Nc] = A[M x K] * B[K x Nc], B given as W^T [Nc x K].
// Block = 128x128 tile, 256 threads (8 waves 4Mx2N), wave tile 32x64:
// 8 WMMAs per wave per 32-wide K step.
// ---------------------------------------------------------------------------
struct EpiArgs {
  float* out_f32;     // mode 0
  const float* bias;  // modes 0,2,3
  __bf16* out_bf;     // mode 2
  int ldo;
  __bf16 *q, *k, *vT; // mode 1
  float* xres;        // mode 3 (residual, fp32)
  const float *mm, *mf;
  int js;             // gate split index (2 = ga_a, 5 = ga_m)
};

template <int MODE>
__device__ __forceinline__ void epi_store(int row, int col, float val,
                                          const EpiArgs& e) {
  if constexpr (MODE == 0) {            // fp32 + bias (modulation GEMMs)
    e.out_f32[(size_t)row * e.ldo + col] = val + e.bias[col];
  } else if constexpr (MODE == 1) {     // QKV scatter; fold 1/sqrt(DH) into Q
    int part = col >> 10, hh = (col >> 6) & 15, d = col & 63;
    int b = row >> 10, nn = row & 1023;
    if (part == 0)
      e.q[((size_t)(b * HH + hh) * NN + nn) * DHH + d] = (__bf16)(val * 0.125f);
    else if (part == 1)
      e.k[((size_t)(b * HH + hh) * NN + nn) * DHH + d] = (__bf16)val;
    else
      e.vT[((size_t)(b * HH + hh) * DHH + d) * NN + nn] = (__bf16)val;
  } else if constexpr (MODE == 2) {     // exact GELU -> bf16 (MLP hidden)
    float t = val + e.bias[col];
    float gl = 0.5f * t * (1.f + erff(t * 0.70710678118654752f));
    e.out_bf[(size_t)row * e.ldo + col] = (__bf16)gl;
  } else {                              // MODE 3: x += gate * (val + bias)
    float gate = (col < CC)
                     ? e.mm[(size_t)row * 6 * CC + e.js * CC + col]
                     : e.mf[(size_t)row * 18 * CC + e.js * 3 * CC + (col - CC)];
    e.xres[(size_t)row * DIMD + col] += gate * (val + e.bias[col]);
  }
}

template <int MODE>
__global__ void k_gemm(const __bf16* __restrict__ A, const __bf16* __restrict__ Bt,
                       int K, int Nc, EpiArgs e) {
  __shared__ __bf16 lA[128][40];   // padded rows (80B) dodge bank conflicts
  __shared__ __bf16 lB[128][40];
  int tid = threadIdx.x;
  int rowBase = blockIdx.y * 128, colBase = blockIdx.x * 128;
  int w = tid >> 5, lane = tid & 31;
  int wM = w & 3, wN = w >> 2;
  int ln = lane & 15, kb = (lane < 16) ? 0 : 8;
  v8f acc[2][4] = {};

  int lr = tid >> 1, lc = (tid & 1) * 16;  // 16 bf16 (two b128) per thread
  const __bf16* ga = A + (size_t)(rowBase + lr) * K + lc;
  const __bf16* gb = Bt + (size_t)(colBase + lr) * K + lc;

  for (int k0 = 0; k0 < K; k0 += 32) {
    *(v8bf*)&lA[lr][lc]     = *(const v8bf*)(ga + k0);
    *(v8bf*)&lA[lr][lc + 8] = *(const v8bf*)(ga + k0 + 8);
    *(v8bf*)&lB[lr][lc]     = *(const v8bf*)(gb + k0);
    *(v8bf*)&lB[lr][lc + 8] = *(const v8bf*)(gb + k0 + 8);
    if (k0 + 32 < K) {                 // global_prefetch_b8 next tile
      __builtin_prefetch(ga + k0 + 32, 0, 1);
      __builtin_prefetch(gb + k0 + 32, 0, 1);
    }
    __syncthreads();
    v16bf a0 = load_frag(&lA[wM * 32 + ln][0], kb);
    v16bf a1 = load_frag(&lA[wM * 32 + 16 + ln][0], kb);
    v16bf b0 = load_frag(&lB[wN * 64 + ln][0], kb);
    v16bf b1 = load_frag(&lB[wN * 64 + 16 + ln][0], kb);
    v16bf b2 = load_frag(&lB[wN * 64 + 32 + ln][0], kb);
    v16bf b3 = load_frag(&lB[wN * 64 + 48 + ln][0], kb);
    acc[0][0] = wmma_bf16(a0, b0, acc[0][0]);
    acc[0][1] = wmma_bf16(a0, b1, acc[0][1]);
    acc[0][2] = wmma_bf16(a0, b2, acc[0][2]);
    acc[0][3] = wmma_bf16(a0, b3, acc[0][3]);
    acc[1][0] = wmma_bf16(a1, b0, acc[1][0]);
    acc[1][1] = wmma_bf16(a1, b1, acc[1][1]);
    acc[1][2] = wmma_bf16(a1, b2, acc[1][2]);
    acc[1][3] = wmma_bf16(a1, b3, acc[1][3]);
    __syncthreads();
  }

#pragma unroll
  for (int mi = 0; mi < 2; mi++) {
    int row0 = rowBase + wM * 32 + mi * 16 + ((lane < 16) ? 0 : 8);
#pragma unroll
    for (int ni = 0; ni < 4; ni++) {
      int col0 = colBase + wN * 64 + ni * 16 + ln;
#pragma unroll
      for (int r = 0; r < 8; r++)
        epi_store<MODE>(row0 + r, col0, acc[mi][ni][r], e);
    }
  }
}

// ---------------------------------------------------------------------------
// Flash attention: one block per (b,h, 64 query rows); 4 waves x 16 rows.
// q (pre-scaled),k row-major [bh][N][DH] bf16; vT transposed [bh][DH][N] bf16.
// 64 keys per iteration: 8 S-WMMAs, 8 O-WMMAs, 2 rowsum-WMMAs (P @ ones),
// DPP-only max reduction (no LDS butterflies).
// ---------------------------------------------------------------------------
__global__ void k_attn(const __bf16* __restrict__ q, const __bf16* __restrict__ k,
                       const __bf16* __restrict__ vT, __bf16* __restrict__ o) {
  __shared__ __bf16 lp[4][16][72];  // per-wave P tile (16x64, padded)
  int w = threadIdx.x >> 5, lane = threadIdx.x & 31;
  int ln = lane & 15, kb = (lane < 16) ? 0 : 8;
  int bh = blockIdx.y;                // b*H + h
  int b = bh >> 4, h = bh & 15;
  int r0 = blockIdx.x * 64 + w * 16;  // query row within sequence

  const __bf16* qbh = q + (size_t)bh * NN * DHH;
  const __bf16* kbh = k + (size_t)bh * NN * DHH;
  const __bf16* vbh = vT + (size_t)bh * DHH * NN;

  const __bf16* qrow = qbh + (size_t)(r0 + ln) * DHH;
  v16bf qa0 = load_frag(qrow, kb);        // d = 0..31
  v16bf qa1 = load_frag(qrow, 32 + kb);   // d = 32..63

  v16bf ones;
#pragma unroll
  for (int i = 0; i < 16; i++) ones[i] = (__bf16)1.f;

  v8f oa[4] = {};      // O accumulators (4 x 16-d tiles)
  v8f lacc = {};       // softmax denominator accumulator = P @ ones
  float mrow[8];
#pragma unroll
  for (int r = 0; r < 8; r++) mrow[r] = -1e30f;

  for (int j = 0; j < NN; j += 64) {
    v8f s[4];
#pragma unroll
    for (int t = 0; t < 4; t++) {
      const __bf16* kr = kbh + (size_t)(j + t * 16 + ln) * DHH;
      v8f z = {};
      s[t] = wmma_bf16(qa1, load_frag(kr, 32 + kb),
                       wmma_bf16(qa0, load_frag(kr, kb), z));
    }
#pragma unroll
    for (int r = 0; r < 8; r++) {
      float rm = fmaxf(fmaxf(s[0][r], s[1][r]), fmaxf(s[2][r], s[3][r]));
      rm = rowmax16(rm);                      // 4 DPP + 4 max, pure VALU
      float mnew = fmaxf(mrow[r], rm);
      float corr = __expf(mrow[r] - mnew);
      mrow[r] = mnew;
      oa[0][r] *= corr; oa[1][r] *= corr; oa[2][r] *= corr; oa[3][r] *= corr;
      lacc[r] *= corr;
      int mrw = (lane < 16) ? r : (r + 8);
#pragma unroll
      for (int t = 0; t < 4; t++)
        lp[w][mrw][t * 16 + ln] = (__bf16)__expf(s[t][r] - mnew);
    }
    __builtin_amdgcn_wave_barrier();  // same-wave DS ops are in-order on CDNA5
    v16bf pa0 = load_frag(&lp[w][ln][0], kb);        // P cols j..j+31
    v16bf pa1 = load_frag(&lp[w][ln][0], 32 + kb);   // P cols j+32..j+63
    lacc = wmma_bf16(pa1, ones, wmma_bf16(pa0, ones, lacc));
#pragma unroll
    for (int dt = 0; dt < 4; dt++) {
      const __bf16* vrow = vbh + (size_t)(dt * 16 + ln) * NN + j;
      oa[dt] = wmma_bf16(pa1, load_frag(vrow + 32, kb),
                         wmma_bf16(pa0, load_frag(vrow, kb), oa[dt]));
    }
  }

  int rbase = r0 + ((lane < 16) ? 0 : 8);
#pragma unroll
  for (int r = 0; r < 8; r++) {
    float inv = 1.f / lacc[r];
    __bf16* orow = o + (size_t)(b * NN + rbase + r) * DIMD + h * DHH;
    orow[ln]      = (__bf16)(oa[0][r] * inv);
    orow[16 + ln] = (__bf16)(oa[1][r] * inv);
    orow[32 + ln] = (__bf16)(oa[2][r] * inv);
    orow[48 + ln] = (__bf16)(oa[3][r] * inv);
  }
}

// ---------------------------------------------------------------------------
// Host orchestration
// ---------------------------------------------------------------------------
extern "C" void kernel_launch(void* const* d_in, const int* in_sizes, int n_in,
                              void* d_out, int out_size, void* d_ws, size_t ws_size,
                              hipStream_t stream) {
  (void)in_sizes; (void)n_in; (void)out_size; (void)ws_size;
  const float* x    = (const float*)d_in[0];
  const float* mu   = (const float*)d_in[1];
  const float* f    = (const float*)d_in[2];
  const float* Wqkv = (const float*)d_in[3];
  const float* Wout = (const float*)d_in[4];
  const float* bout = (const float*)d_in[5];
  const float* W1   = (const float*)d_in[6];
  const float* b1   = (const float*)d_in[7];
  const float* W2   = (const float*)d_in[8];
  const float* b2   = (const float*)d_in[9];
  const float* Wmu  = (const float*)d_in[10];
  const float* bmu  = (const float*)d_in[11];
  const float* Wf   = (const float*)d_in[12];
  const float* bf_  = (const float*)d_in[13];
  const float* g1   = (const float*)d_in[14];
  const float* be1  = (const float*)d_in[15];
  const float* g2   = (const float*)d_in[16];
  const float* be2  = (const float*)d_in[17];
  float* xout = (float*)d_out;  // residual lives in d_out

  char* ws = (char*)d_ws;
  size_t off = 0;
  auto alloc = [&](size_t bytes) -> void* {
    void* p = ws + off;
    off = (off + bytes + 255) & ~(size_t)255;
    return p;
  };
  __bf16* smu = (__bf16*)alloc((size_t)MTOT * CC * 2);
  __bf16* sf  = (__bf16*)alloc((size_t)MTOT * CC * 2);
  __bf16* wT  = (__bf16*)alloc((size_t)MLPDIM * DIMD * 2);   // reused per GEMM
  float*  mmv = (float*)alloc((size_t)MTOT * 6 * CC * 4);
  float*  mfv = (float*)alloc((size_t)MTOT * 18 * CC * 4);
  __bf16* hb  = (__bf16*)alloc((size_t)MTOT * DIMD * 2);     // h1 / o / h2
  __bf16* qb  = (__bf16*)alloc((size_t)BB * HH * NN * DHH * 2);
  __bf16* kbq = (__bf16*)alloc((size_t)BB * HH * NN * DHH * 2);
  __bf16* vTb = (__bf16*)alloc((size_t)BB * HH * NN * DHH * 2);
  __bf16* gb  = (__bf16*)alloc((size_t)MTOT * MLPDIM * 2);

  hipMemcpyAsync(xout, x, (size_t)MTOT * DIMD * 4, hipMemcpyDeviceToDevice, stream);
  k_silu_bf16<<<(MTOT * CC + 255) / 256, 256, 0, stream>>>(mu, smu, MTOT * CC);
  k_silu_bf16<<<(MTOT * CC + 255) / 256, 256, 0, stream>>>(f, sf, MTOT * CC);

  for (int i = 0; i < LLAYERS; i++) {
    const float* Wmu_i  = Wmu + (size_t)i * CC * 6 * CC;
    const float* Wf_i   = Wf + (size_t)i * CC * 18 * CC;
    const float* Wqkv_i = Wqkv + (size_t)i * DIMD * 3 * HH * DHH;
    const float* Wout_i = Wout + (size_t)i * HH * DHH * DIMD;
    const float* W1_i   = W1 + (size_t)i * DIMD * MLPDIM;
    const float* W2_i   = W2 + (size_t)i * MLPDIM * DIMD;

    // --- modulation GEMMs: mm = silu(mu) @ Wmu + bmu ; mf = silu(f) @ Wf + bf
    k_transpose_cast<<<dim3(6 * CC / 32, CC / 32), 256, 0, stream>>>(Wmu_i, wT, CC, 6 * CC);
    {
      EpiArgs e{}; e.out_f32 = mmv; e.bias = bmu + (size_t)i * 6 * CC; e.ldo = 6 * CC;
      k_gemm<0><<<dim3(6 * CC / 128, MTOT / 128), 256, 0, stream>>>(smu, wT, CC, 6 * CC, e);
    }
    k_transpose_cast<<<dim3(18 * CC / 32, CC / 32), 256, 0, stream>>>(Wf_i, wT, CC, 18 * CC);
    {
      EpiArgs e{}; e.out_f32 = mfv; e.bias = bf_ + (size_t)i * 18 * CC; e.ldo = 18 * CC;
      k_gemm<0><<<dim3(18 * CC / 128, MTOT / 128), 256, 0, stream>>>(sf, wT, CC, 18 * CC, e);
    }

    // --- attention branch: h1 = LN(x)*(1+sc_a)+sh_a ; qkv; flash; outproj+gate
    k_ln_mod<<<MTOT, 256, 0, stream>>>(xout, g1 + (size_t)i * DIMD,
                                       be1 + (size_t)i * DIMD, mmv, mfv,
                                       /*jsh=*/0, /*jsc=*/1, hb);
    k_transpose_cast<<<dim3(3 * HH * DHH / 32, DIMD / 32), 256, 0, stream>>>(
        Wqkv_i, wT, DIMD, 3 * HH * DHH);
    {
      EpiArgs e{}; e.q = qb; e.k = kbq; e.vT = vTb;
      k_gemm<1><<<dim3(3 * HH * DHH / 128, MTOT / 128), 256, 0, stream>>>(
          hb, wT, DIMD, 3 * HH * DHH, e);
    }
    k_attn<<<dim3(NN / 64, BB * HH), 128, 0, stream>>>(qb, kbq, vTb, hb);
    k_transpose_cast<<<dim3(DIMD / 32, DIMD / 32), 256, 0, stream>>>(Wout_i, wT, DIMD, DIMD);
    {
      EpiArgs e{}; e.xres = xout; e.bias = bout + (size_t)i * DIMD;
      e.mm = mmv; e.mf = mfv; e.js = 2;  // ga_a
      k_gemm<3><<<dim3(DIMD / 128, MTOT / 128), 256, 0, stream>>>(hb, wT, DIMD, DIMD, e);
    }

    // --- MLP branch: h2 = LN(x)*(1+sc_m)+sh_m ; gelu(h2@W1+b1)@W2+b2, gated
    k_ln_mod<<<MTOT, 256, 0, stream>>>(xout, g2 + (size_t)i * DIMD,
                                       be2 + (size_t)i * DIMD, mmv, mfv,
                                       /*jsh=*/3, /*jsc=*/4, hb);
    k_transpose_cast<<<dim3(MLPDIM / 32, DIMD / 32), 256, 0, stream>>>(W1_i, wT, DIMD, MLPDIM);
    {
      EpiArgs e{}; e.out_bf = gb; e.bias = b1 + (size_t)i * MLPDIM; e.ldo = MLPDIM;
      k_gemm<2><<<dim3(MLPDIM / 128, MTOT / 128), 256, 0, stream>>>(hb, wT, DIMD, MLPDIM, e);
    }
    k_transpose_cast<<<dim3(DIMD / 32, MLPDIM / 32), 256, 0, stream>>>(W2_i, wT, MLPDIM, DIMD);
    {
      EpiArgs e{}; e.xres = xout; e.bias = b2 + (size_t)i * DIMD;
      e.mm = mmv; e.mf = mfv; e.js = 5;  // ga_m
      k_gemm<3><<<dim3(DIMD / 128, MTOT / 128), 256, 0, stream>>>(gb, wT, MLPDIM, DIMD, e);
    }
  }
}
